// StatelessRNN_53996328845611
// MI455X (gfx1250) — matrix operand order repaired and checked by reading, not compiled, not run on hardware
//
#include <hip/hip_runtime.h>
#include <hip/hip_bf16.h>

// ---------------------------------------------------------------------------
// Stacked GRU inference for MI455X (gfx1250, wave32, WMMA, async-LDS, TDM).
//   B=64, T=512, F=64, E=128, H=512, 3H=1536, I0=F+E=192.
// Pipeline:
//   1) convert weights to f16, transposed (N-major) for WMMA B-fragments
//   2) build seq0 = concat(x, emb[ids]) in f16
//   3) xproj0 = seq0 @ W0 + b0[0]   (WMMA GEMM, register double-buffered)
//   4) persistent GRU scan layer 0: U slice staged to LDS once via the
//      Tensor Data Mover (3 descriptors, padded LDS pitch), h state staged
//      per step via global_load_async_to_lds_b128, 1 grid barrier/step
//   5) xproj1 = seq1 @ W1 + b1[0]
//   6) persistent GRU scan layer 1 -> final h (f32)
//   7) out = sigmoid(h @ Wo + bo)
// ---------------------------------------------------------------------------

typedef _Float16 v16h __attribute__((ext_vector_type(16)));
typedef _Float16 v8h  __attribute__((ext_vector_type(8)));
typedef float    v8f  __attribute__((ext_vector_type(8)));
typedef unsigned int u32x4 __attribute__((ext_vector_type(4)));
typedef unsigned int u32x8 __attribute__((ext_vector_type(8)));

#define BATCH  64
#define TSEQ   512
#define IN0    192      // F + E
#define HDIM   512
#define G3H    1536     // 3*H
#define SCAN_BLOCKS 16  // persistent workgroups in the scan kernel

// LDS layout for the scan kernel (dynamic shared memory, base offset 0):
// rows padded to 520 halfs = 1040 B (== 16 mod 256 -> fragment lanes hit
// all 64 LDS banks).
#define U_LDS_STRIDE 520
#define US_BYTES (96 * U_LDS_STRIDE * 2)   // 99840  : U slice, 96 x 512 f16
#define HS_OFF   US_BYTES
#define HS_BYTES (64 * U_LDS_STRIDE * 2)   // 66560  : h stage, 64 x 512 f16
#define REC_OFF  (HS_OFF + HS_BYTES)       // 166400 : rec tile 64 x 100 f32
#define SCAN_LDS (REC_OFF + 64 * 100 * 4)  // 192000 B total (< 320 KB)

// ---------------- WMMA helpers (CDNA5 16x16x32 f16 -> f32) -----------------

__device__ __forceinline__ v8f wmma_f16(v16h a, v16h b, v8f c) {
  // (neg_a, A, neg_b, B, c_mod, C, reuse_a, reuse_b)
  return __builtin_amdgcn_wmma_f32_16x16x32_f16(false, a, false, b,
                                                (short)0, c, false, false);
}

// A fragment: 16x32 f16 tile of row-major A (ld halfs per row).
// Lane L: row m0+(L&15); halfs 0..7 -> K = k0 + 8*(L>>4) + i,
//                        halfs 8..15 -> K = k0 + 16 + 8*(L>>4) + i.
__device__ __forceinline__ v16h load_a_frag(const _Float16* __restrict__ A,
                                            int ld, int m0, int k0, int lane) {
  const _Float16* p = A + (size_t)(m0 + (lane & 15)) * ld
                        + (k0 + ((lane >> 4) << 3));
  v8h lo = *(const v8h*)(p);
  v8h hi = *(const v8h*)(p + 16);
  v16h r;
#pragma unroll
  for (int i = 0; i < 8; ++i) { r[i] = lo[i]; r[i + 8] = hi[i]; }
  return r;
}

// B fragment: 32x16 f16 tile with B stored transposed (N-major: Bt[n][k]).
// Lane L: col n0+(L&15); halfs 0..7 -> K = k0 + 8*(L>>4) + i,
//                        halfs 8..15 -> K = k0 + 16 + 8*(L>>4) + i.
__device__ __forceinline__ v16h load_b_fragT(const _Float16* __restrict__ Bt,
                                             int ld, int n0, int k0, int lane) {
  const _Float16* p = Bt + (size_t)(n0 + (lane & 15)) * ld
                         + (k0 + ((lane >> 4) << 3));
  v8h lo = *(const v8h*)(p);
  v8h hi = *(const v8h*)(p + 16);
  v16h r;
#pragma unroll
  for (int i = 0; i < 8; ++i) { r[i] = lo[i]; r[i + 8] = hi[i]; }
  return r;
}

// Same fragment loaders but sourced from the padded LDS stages.
__device__ __forceinline__ v16h lds_a_frag(const _Float16* S,
                                           int m0, int k0, int lane) {
  const _Float16* p = S + (m0 + (lane & 15)) * U_LDS_STRIDE
                        + (k0 + ((lane >> 4) << 3));
  v8h lo = *(const v8h*)(p);
  v8h hi = *(const v8h*)(p + 16);
  v16h r;
#pragma unroll
  for (int i = 0; i < 8; ++i) { r[i] = lo[i]; r[i + 8] = hi[i]; }
  return r;
}

__device__ __forceinline__ v16h lds_b_frag(const _Float16* S,
                                           int lc0, int k0, int lane) {
  const _Float16* p = S + (lc0 + (lane & 15)) * U_LDS_STRIDE
                        + (k0 + ((lane >> 4) << 3));
  v8h lo = *(const v8h*)(p);
  v8h hi = *(const v8h*)(p + 16);
  v16h r;
#pragma unroll
  for (int i = 0; i < 8; ++i) { r[i] = lo[i]; r[i + 8] = hi[i]; }
  return r;
}

__device__ __forceinline__ float sigmoidf_fast(float x) {
  return 1.0f / (1.0f + __expf(-x));
}

// Issue one TDM 2-D tile load Global->LDS (wave-level; EXEC-independent).
//   tile: rows x 512 f16 elements, source row stride 512 elements,
//   LDS pitch 1040 B via pad_enable (16 B pad after every 1024 B).
__device__ __forceinline__ void tdm_load_tile(const _Float16* gsrc,
                                              unsigned lds_byte_addr,
                                              unsigned rows) {
  unsigned long long ga = (unsigned long long)(const void*)gsrc;
  u32x4 g0;
  g0[0] = 1u;                                   // count=1, user descriptor
  g0[1] = lds_byte_addr;                        // LDS byte address
  g0[2] = (unsigned)(ga & 0xffffffffu);         // global addr [31:0]
  g0[3] = (unsigned)((ga >> 32) & 0x01ffffffu)  // global addr [56:32]
        | (2u << 30);                           // type = 2 ("image")
  u32x8 g1;
  g1[0] = (1u << 16)        // data_size = 2 bytes
        | (1u << 20)        // pad_enable
        | (7u << 22)        // pad_interval: 256 DWORDs = 1024 B
        | (3u << 25);       // pad_amount:   4 DWORDs  = 16 B
  g1[1] = (512u << 16);     // tensor_dim0 = 512 (low 16 bits)
  g1[2] = (rows << 16);     // tensor_dim0 hi=0 | tensor_dim1 = rows
  g1[3] = (512u << 16);     // tensor_dim1 hi=0 | tile_dim0 = 512
  g1[4] = rows;             // tile_dim1 = rows, tile_dim2 = 0 (2-D)
  g1[5] = 512u;             // tensor_dim0_stride = 512 (low 32)
  g1[6] = 0u;               // stride hi | tensor_dim1_stride lo
  g1[7] = 0u;
  u32x4 g2 = {0u, 0u, 0u, 0u};
  u32x4 g3 = {0u, 0u, 0u, 0u};
  asm volatile("tensor_load_to_lds %0, %1, %2, %3"
               :: "s"(g0), "s"(g1), "s"(g2), "s"(g3) : "memory");
}

// ---------------- small setup kernels --------------------------------------

__global__ void k_init_ctrs(unsigned* c) {
  if (threadIdx.x < 64) c[threadIdx.x] = 0u;
}

// src: K x N row-major f32  ->  dst: N x K row-major f16 (transpose+convert)
__global__ void k_cvt_transpose(const float* __restrict__ src,
                                _Float16* __restrict__ dst, int K, int N) {
  int i = blockIdx.x * blockDim.x + threadIdx.x;
  if (i >= K * N) return;
  int n = i % N, k = i / N;                       // coalesced read
  dst[(size_t)n * K + k] = (_Float16)src[i];
}

// seq0[r][0:64] = x[r], seq0[r][64:192] = emb[ids[r]]
__global__ void k_build_seq(const float* __restrict__ x,
                            const int* __restrict__ ids,
                            const float* __restrict__ emb,
                            _Float16* __restrict__ seq) {
  int r = blockIdx.x;          // b*T + t
  int c = threadIdx.x;         // 0..191
  float v;
  if (c < 64) v = x[(size_t)r * 64 + c];
  else        v = emb[(size_t)ids[r] * 128 + (c - 64)];
  seq[(size_t)r * IN0 + c] = (_Float16)v;
}

// ---------------- big input-projection GEMM --------------------------------
// C(MxN,f16) = A(MxK,f16) @ B(KxN) + bias, B pre-transposed (Bt: NxK).
// Block = 256 threads (8 waves), 128x64 macro tile; wave w -> M-subtile w*16,
// 4 N-subtiles. Fragments register double-buffered so global-load latency
// overlaps the 4 WMMAs of the previous K-slice.
__global__ void __launch_bounds__(256)
k_gemm_bias_f16(const _Float16* __restrict__ A,
                const _Float16* __restrict__ Bt,
                const float* __restrict__ bias,
                _Float16* __restrict__ C,
                int M, int N, int K) {
  const int lane = threadIdx.x & 31;
  const int w    = threadIdx.x >> 5;
  const int m0   = blockIdx.x * 128 + w * 16;
  const int n0   = blockIdx.y * 64;

  v8f acc0 = {}, acc1 = {}, acc2 = {}, acc3 = {};

  v16h a  = load_a_frag(A, K, m0, 0, lane);
  v16h b0 = load_b_fragT(Bt, K, n0 +  0, 0, lane);
  v16h b1 = load_b_fragT(Bt, K, n0 + 16, 0, lane);
  v16h b2 = load_b_fragT(Bt, K, n0 + 32, 0, lane);
  v16h b3 = load_b_fragT(Bt, K, n0 + 48, 0, lane);

  for (int k0 = 0; k0 < K; k0 += 32) {
    const int kn = (k0 + 32 < K) ? (k0 + 32) : 0;   // wrap: harmless reload
    v16h an  = load_a_frag(A, K, m0, kn, lane);
    v16h bn0 = load_b_fragT(Bt, K, n0 +  0, kn, lane);
    v16h bn1 = load_b_fragT(Bt, K, n0 + 16, kn, lane);
    v16h bn2 = load_b_fragT(Bt, K, n0 + 32, kn, lane);
    v16h bn3 = load_b_fragT(Bt, K, n0 + 48, kn, lane);
    acc0 = wmma_f16(a, b0, acc0);
    acc1 = wmma_f16(a, b1, acc1);
    acc2 = wmma_f16(a, b2, acc2);
    acc3 = wmma_f16(a, b3, acc3);
    a = an; b0 = bn0; b1 = bn1; b2 = bn2; b3 = bn3;
  }
  // Epilogue: D layout -> lane holds col n=(lane&15), rows m = 8*(lane>>4)+v.
  const int nl   = lane & 15;
  const int mrow = m0 + (lane >> 4) * 8;
  v8f* accs[4] = { &acc0, &acc1, &acc2, &acc3 };
#pragma unroll
  for (int j = 0; j < 4; ++j) {
    int n = n0 + j * 16 + nl;
    float bs = bias[n];
#pragma unroll
    for (int v = 0; v < 8; ++v)
      C[(size_t)(mrow + v) * N + n] = (_Float16)((*accs[j])[v] + bs);
  }
}

// ---------------- grid-wide barrier for the persistent scan ----------------

__device__ __forceinline__ void grid_barrier(unsigned* ctr) {
  __threadfence();
  __syncthreads();
  if (threadIdx.x == 0) {
    unsigned old = __hip_atomic_fetch_add(ctr, 1u, __ATOMIC_ACQ_REL,
                                          __HIP_MEMORY_SCOPE_AGENT);
    unsigned target = (old / SCAN_BLOCKS + 1u) * SCAN_BLOCKS;
    while (__hip_atomic_load(ctr, __ATOMIC_ACQUIRE,
                             __HIP_MEMORY_SCOPE_AGENT) < target) {
      __builtin_amdgcn_s_sleep(1);
    }
  }
  __syncthreads();
}

// ---------------- persistent GRU scan --------------------------------------
// Grid = 16 WGs x 256 threads. WG g owns hidden units [g*32, g*32+32).
// One-time: TDM-stage the WG's 96 U columns (96 KB f16) into LDS
//   (3 tensor_load_to_lds descriptors, padded LDS pitch, TENSORcnt).
// Per step t:
//   async-stage h (64x512 f16, 64 KB) into LDS via
//   global_load_async_to_lds_b128 (ASYNCcnt),
//   rec(64x96) = h @ U from LDS fragments: 24 WMMA tiles, 3 per wave,
//   rec -> LDS, gate math, h ping-pong write, one grid barrier.
__global__ void __launch_bounds__(256)
k_gru_scan(const _Float16* __restrict__ xp,    // (B*T) x 1536 preactivations
           const _Float16* __restrict__ Ut,    // 1536 x 512 (N-major)
           const float*    __restrict__ brec,  // recurrent bias, 1536
           float*          __restrict__ hbuf,  // 64 x 512 f32 state
           _Float16*       __restrict__ h16,   // 2 x 64 x 512 f16 ping-pong
           _Float16*       __restrict__ seq_out, // (B*T) x 512 or nullptr
           unsigned*       __restrict__ ctr,
           int T) {
  extern __shared__ char smem[];
  _Float16* Us = (_Float16*)smem;                         // [96][520] f16
  _Float16* hS = (_Float16*)(smem + HS_OFF);              // [64][520] f16
  float (*recS)[100] = (float (*)[100])(smem + REC_OFF);  // [64][100] f32

  const int g    = blockIdx.x;                  // unit group
  const int tid  = threadIdx.x;
  const int lane = tid & 31;
  const int w    = tid >> 5;
  const int mt   = w & 3;                       // M tile (batch rows mt*16..)
  const int nb   = (w >> 2) * 3;                // first of 3 N tiles

  // ---- one-time TDM stage: WG's 96 U columns -> LDS (3 gate blocks) ----
  if (w == 0) {                                 // one wave issues the DMAs
#pragma unroll
    for (int gate = 0; gate < 3; ++gate) {
      const _Float16* gsrc = Ut + ((size_t)gate * HDIM + g * 32) * HDIM;
      unsigned lds_addr = (unsigned)(gate * 32 * (U_LDS_STRIDE * 2));
      tdm_load_tile(gsrc, lds_addr, 32u);
    }
  }

  // h0 = 0 (this WG's 32 columns of master + ping buffer 0)
  for (int e = tid; e < BATCH * 32; e += 256) {
    int row = e >> 5, j = g * 32 + (e & 31);
    hbuf[row * HDIM + j] = 0.0f;
    h16[row * HDIM + j] = (_Float16)0.0f;
  }
  if (w == 0) __builtin_amdgcn_s_wait_tensorcnt((short)0);
  grid_barrier(ctr);   // TDM data + h0 visible to all waves / WGs

  for (int t = 0; t < T; ++t) {
    const _Float16* hin  = h16 + (size_t)(t & 1) * (BATCH * HDIM);
    _Float16*       hout = h16 + (size_t)((t + 1) & 1) * (BATCH * HDIM);

    // ---- async stage current h (64x512 f16) -> LDS ----
    for (int q = tid; q < 64 * 64; q += 256) {  // 16 wave-instructions/wave
      int row = q >> 6, chunk = q & 63;
      const _Float16* gp = hin + row * HDIM + chunk * 8;
      unsigned lo = (unsigned)(HS_OFF + row * (U_LDS_STRIDE * 2) + chunk * 16);
      asm volatile("global_load_async_to_lds_b128 %0, %1, off"
                   :: "v"(lo), "v"(gp) : "memory");
    }

    // prefetch next step's x-projections for this WG's columns
    if (t + 1 < T) {
      int row = tid >> 2, part = tid & 3;
      if (part < 3) {
        const _Float16* p = xp + ((size_t)row * T + (t + 1)) * G3H
                               + part * HDIM + g * 32;
        __builtin_prefetch((const void*)p, 0, 1);
      }
    }

    asm volatile("s_wait_asynccnt 0" ::: "memory");
    __syncthreads();   // hS visible to all waves

    // ---- recurrent GEMM from LDS: rec = h @ U (WG's 96 columns) ----
    v8f acc0 = {}, acc1 = {}, acc2 = {};
#pragma unroll 2
    for (int k0 = 0; k0 < HDIM; k0 += 32) {
      v16h a  = lds_a_frag(hS, mt * 16, k0, lane);
      v16h b0 = lds_b_frag(Us, (nb + 0) * 16, k0, lane);
      v16h b1 = lds_b_frag(Us, (nb + 1) * 16, k0, lane);
      v16h b2 = lds_b_frag(Us, (nb + 2) * 16, k0, lane);
      acc0 = wmma_f16(a, b0, acc0);
      acc1 = wmma_f16(a, b1, acc1);
      acc2 = wmma_f16(a, b2, acc2);
    }
    // D fragment -> LDS rec tile (local col = nt*16 + (lane&15))
    {
      const int nl = lane & 15;
      const int mbase = mt * 16 + (lane >> 4) * 8;
#pragma unroll
      for (int v = 0; v < 8; ++v) {
        recS[mbase + v][(nb + 0) * 16 + nl] = acc0[v];
        recS[mbase + v][(nb + 1) * 16 + nl] = acc1[v];
        recS[mbase + v][(nb + 2) * 16 + nl] = acc2[v];
      }
    }
    __syncthreads();

    // ---- gate math: 64 rows x 32 units, 8 per thread ----
    for (int e = tid; e < BATCH * 32; e += 256) {
      int row = e >> 5, u = e & 31;
      int j = g * 32 + u;
      float rz = recS[row][u]      + brec[j];
      float rr = recS[row][32 + u] + brec[HDIM + j];
      float rh = recS[row][64 + u] + brec[2 * HDIM + j];
      const _Float16* xrow = xp + ((size_t)row * T + t) * G3H;
      float xz = (float)xrow[j];
      float xr = (float)xrow[HDIM + j];
      float xh = (float)xrow[2 * HDIM + j];
      float h  = hbuf[row * HDIM + j];
      float z  = sigmoidf_fast(xz + rz);
      float r  = sigmoidf_fast(xr + rr);
      float hh = tanhf(xh + r * rh);              // reset_after candidate
      float hn = z * h + (1.0f - z) * hh;
      hbuf[row * HDIM + j] = hn;
      hout[row * HDIM + j] = (_Float16)hn;
      if (seq_out)
        seq_out[((size_t)row * T + t) * HDIM + j] = (_Float16)hn;
    }
    grid_barrier(ctr);   // writes of step t visible before step t+1 reads
  }
}

// ---------------- output head ----------------------------------------------

__global__ void k_head(const float* __restrict__ hbuf,
                       const float* __restrict__ Wo,
                       const float* __restrict__ bo,
                       float* __restrict__ out) {
  int b = threadIdx.x;
  if (b >= BATCH) return;
  float s = bo[0];
  for (int k = 0; k < HDIM; ++k) s += hbuf[b * HDIM + k] * Wo[k];
  out[b] = 1.0f / (1.0f + __expf(-s));
}

// ---------------- launch ----------------------------------------------------

extern "C" void kernel_launch(void* const* d_in, const int* in_sizes, int n_in,
                              void* d_out, int out_size, void* d_ws, size_t ws_size,
                              hipStream_t stream) {
  (void)in_sizes; (void)n_in; (void)out_size; (void)ws_size;
  const float* x   = (const float*)d_in[0];
  const int*   ids = (const int*)d_in[1];
  const float* emb = (const float*)d_in[2];
  const float* W0  = (const float*)d_in[3];
  const float* U0  = (const float*)d_in[4];
  const float* b0  = (const float*)d_in[5];
  const float* W1  = (const float*)d_in[6];
  const float* U1  = (const float*)d_in[7];
  const float* b1  = (const float*)d_in[8];
  const float* Wo  = (const float*)d_in[9];
  const float* bo  = (const float*)d_in[10];
  float* out = (float*)d_out;

  char* ws = (char*)d_ws;
  size_t off = 0;
  auto carve = [&](size_t bytes) -> char* {
    char* p = ws + off;
    off += (bytes + 255) & ~(size_t)255;
    return p;
  };
  _Float16* seq0 = (_Float16*)carve((size_t)BATCH * TSEQ * IN0 * 2);
  _Float16* W0t  = (_Float16*)carve((size_t)G3H * IN0 * 2);
  _Float16* U0t  = (_Float16*)carve((size_t)G3H * HDIM * 2);
  _Float16* W1t  = (_Float16*)carve((size_t)G3H * HDIM * 2);
  _Float16* U1t  = (_Float16*)carve((size_t)G3H * HDIM * 2);
  _Float16* xp0  = (_Float16*)carve((size_t)BATCH * TSEQ * G3H * 2);
  _Float16* xp1  = (_Float16*)carve((size_t)BATCH * TSEQ * G3H * 2);
  _Float16* seq1 = (_Float16*)carve((size_t)BATCH * TSEQ * HDIM * 2);
  float*    hbuf = (float*)carve((size_t)BATCH * HDIM * 4);
  _Float16* h16  = (_Float16*)carve((size_t)2 * BATCH * HDIM * 2);
  unsigned* ctrs = (unsigned*)carve(256);

  const int M = BATCH * TSEQ;  // 32768

  hipLaunchKernelGGL(k_init_ctrs, dim3(1), dim3(64), 0, stream, ctrs);

  hipLaunchKernelGGL(k_cvt_transpose, dim3((IN0 * G3H + 255) / 256), dim3(256),
                     0, stream, W0, W0t, IN0, G3H);
  hipLaunchKernelGGL(k_cvt_transpose, dim3((HDIM * G3H + 255) / 256), dim3(256),
                     0, stream, U0, U0t, HDIM, G3H);
  hipLaunchKernelGGL(k_cvt_transpose, dim3((HDIM * G3H + 255) / 256), dim3(256),
                     0, stream, W1, W1t, HDIM, G3H);
  hipLaunchKernelGGL(k_cvt_transpose, dim3((HDIM * G3H + 255) / 256), dim3(256),
                     0, stream, U1, U1t, HDIM, G3H);

  hipLaunchKernelGGL(k_build_seq, dim3(M), dim3(IN0), 0, stream,
                     x, ids, emb, seq0);

  hipLaunchKernelGGL(k_gemm_bias_f16, dim3(M / 128, G3H / 64), dim3(256), 0,
                     stream, seq0, W0t, b0 /*row 0*/, xp0, M, G3H, IN0);

  hipLaunchKernelGGL(k_gru_scan, dim3(SCAN_BLOCKS), dim3(256), SCAN_LDS,
                     stream, xp0, U0t, b0 + G3H /*row 1*/, hbuf, h16, seq1,
                     ctrs + 0, TSEQ);

  hipLaunchKernelGGL(k_gemm_bias_f16, dim3(M / 128, G3H / 64), dim3(256), 0,
                     stream, seq1, W1t, b1 /*row 0*/, xp1, M, G3H, HDIM);

  hipLaunchKernelGGL(k_gru_scan, dim3(SCAN_BLOCKS), dim3(256), SCAN_LDS,
                     stream, xp1, U1t, b1 + G3H /*row 1*/, hbuf, h16,
                     (_Float16*)nullptr, ctrs + 32, TSEQ);

  hipLaunchKernelGGL(k_head, dim3(1), dim3(64), 0, stream, hbuf, Wo, bo, out);
}
